// MultiHead_TempAttention_14894946583436
// MI455X (gfx1250) — matrix operand order repaired and checked by reading, not compiled, not run on hardware
//
#include <hip/hip_runtime.h>
#include <math.h>

// Problem constants
#define Bn    8
#define Tn    1024
#define Dm    512
#define Hn    8
#define DKv   64
#define KF    3
#define CQ    512          // DK * H
#define NROWS (Bn * Tn)    // 8192
#define OUT0  ((size_t)Bn * Tn * Dm)            // 4,194,304 floats (out)
#define LN_EPS 1e-5f

typedef __attribute__((ext_vector_type(16))) __bf16 bf16x16;
typedef __attribute__((ext_vector_type(8)))  float  f32x8;

struct U4 { unsigned int a, b, c, d; };

union Frag {
  bf16x16 v;
  __bf16 h[16];
  unsigned short u[16];
  U4 q[2];
};

// Hardware f32 -> bf16 (v_cvt_pk_bf16_f32 / v_cvt path, no bit-twiddle VALU)
__device__ __forceinline__ __bf16 tobf(float f) { return (__bf16)f; }

__device__ __forceinline__ void pack8(Frag& a, int base, float4 f0, float4 f1) {
  a.h[base+0] = tobf(f0.x); a.h[base+1] = tobf(f0.y);
  a.h[base+2] = tobf(f0.z); a.h[base+3] = tobf(f0.w);
  a.h[base+4] = tobf(f1.x); a.h[base+5] = tobf(f1.y);
  a.h[base+6] = tobf(f1.z); a.h[base+7] = tobf(f1.w);
}

__device__ __forceinline__ f32x8 wmma_bf16(const Frag& a, const Frag& b, f32x8 c) {
  return __builtin_amdgcn_wmma_f32_16x16x32_bf16(false, a.v, false, b.v,
                                                 (short)0, c, false, false);
}

// ---------------------------------------------------------------------------
// Kernel 0: repack weights to bf16, B-fragment friendly (N-major rows, K contiguous)
//   wq_t/wk_t : [c][kk*512 + cin]   (c = conv out-channel, K-dim = 1536)
//   wv_t/wfc_t: [cout][cin]         (transpose of the [cin][cout] fp32 weights)
// ---------------------------------------------------------------------------
__global__ void prep_weights(const float* __restrict__ Wq, const float* __restrict__ Wk,
                             const float* __restrict__ Wv, const float* __restrict__ Wfc,
                             __bf16* __restrict__ wq_t, __bf16* __restrict__ wk_t,
                             __bf16* __restrict__ wv_t, __bf16* __restrict__ wfc_t) {
  int idx = blockIdx.x * 256 + threadIdx.x;
  if (idx < CQ * Dm * KF) {                       // 786432
    int c   = idx / (Dm * KF);
    int rem = idx % (Dm * KF);
    int kk  = rem >> 9;
    int cin = rem & 511;
    int s   = c * (Dm * KF) + cin * KF + kk;      // (C_out, C_in, K) row-major
    wq_t[idx] = tobf(Wq[s]);
    wk_t[idx] = tobf(Wk[s]);
  }
  if (idx < Dm * Dm) {                            // 262144
    int co = idx >> 9;
    int ci = idx & 511;
    wv_t[idx]  = tobf(Wv [ci * Dm + co]);
    wfc_t[idx] = tobf(Wfc[ci * Dm + co]);
  }
}

// ---------------------------------------------------------------------------
// Kernel 1: causal-conv projection for Q and K as one GEMM, K-dim = 3*512.
//   out[bh][t][d] (bf16), channel c = d*8 + h.  blockIdx.z selects Q/K.
//   Boundary (t < 2) handling hoisted to a wave-uniform branch: only the
//   ttile==0 tiles can touch the causal left pad.
// ---------------------------------------------------------------------------
__global__ void qk_conv_proj(const float* __restrict__ qin, const float* __restrict__ kin,
                             const __bf16* __restrict__ wq_t,
                             const __bf16* __restrict__ wk_t,
                             const float* __restrict__ bq, const float* __restrict__ bk,
                             __bf16* __restrict__ qh, __bf16* __restrict__ kh) {
  const int wid  = threadIdx.x >> 5;
  const int lane = threadIdx.x & 31;
  const int tile = blockIdx.x * 4 + wid;          // 0..2047
  const int ctile = tile & 31;
  const int ttile = tile >> 5;
  const int b   = blockIdx.y;
  const int sel = blockIdx.z;

  const float* src = sel ? kin : qin;
  const __bf16* wt = sel ? wk_t : wq_t;
  const float* bias = sel ? bk : bq;
  __bf16* dst = sel ? kh : qh;

  const int m    = lane & 15;
  const int half = lane >> 4;
  const int tbase = ttile * 16;
  const int cbase = ctile * 16;
  const int t = tbase + m;
  const int c = cbase + m;                        // B-frag column for this lane

  f32x8 acc = {0.f,0.f,0.f,0.f,0.f,0.f,0.f,0.f};
  const __bf16* wrow = wt + (size_t)c * (Dm * KF);

  if (ttile != 0) {
    // interior: t >= 16, every tap row is valid -> branch-free hot loop
    for (int K0 = 0; K0 < Dm * KF; K0 += 32) {
      const int kk   = K0 >> 9;
      const int cin0 = K0 & 511;
      Frag a;
      const float* p = src + ((size_t)(b * Tn + t - 2 + kk)) * Dm + cin0 + half * 8;
      float4 f0 = *(const float4*)p;
      float4 f1 = *(const float4*)(p + 4);
      float4 f2 = *(const float4*)(p + 16);
      float4 f3 = *(const float4*)(p + 20);
      pack8(a, 0, f0, f1);
      pack8(a, 8, f2, f3);
      Frag bf;
      const U4* bp = (const U4*)(wrow + K0 + half * 8);
      bf.q[0] = bp[0]; bf.q[1] = bp[2];
      acc = wmma_bf16(a, bf, acc);
    }
  } else {
    for (int K0 = 0; K0 < Dm * KF; K0 += 32) {
      const int kk   = K0 >> 9;
      const int cin0 = K0 & 511;
      Frag a;
      const int sr = t - 2 + kk;                  // causal left pad
      if (sr >= 0) {
        const float* p = src + ((size_t)(b * Tn + sr)) * Dm + cin0 + half * 8;
        float4 f0 = *(const float4*)p;
        float4 f1 = *(const float4*)(p + 4);
        float4 f2 = *(const float4*)(p + 16);
        float4 f3 = *(const float4*)(p + 20);
        pack8(a, 0, f0, f1);
        pack8(a, 8, f2, f3);
      } else {
        a.q[0] = U4{0,0,0,0}; a.q[1] = U4{0,0,0,0};
      }
      Frag bf;
      const U4* bp = (const U4*)(wrow + K0 + half * 8);
      bf.q[0] = bp[0]; bf.q[1] = bp[2];
      acc = wmma_bf16(a, bf, acc);
    }
  }

  const float bi = bias[c];
  const int d  = c >> 3;
  const int hh = c & 7;
  __bf16* drow = dst + d;
  const size_t bh = (size_t)(b * Hn + hh);
#pragma unroll
  for (int r = 0; r < 8; ++r) {
    int tr = tbase + r + 8 * half;
    drow[(bh * Tn + tr) * DKv] = tobf(acc[r] + bi);
  }
}

// ---------------------------------------------------------------------------
// Kernel 2: V projection (v @ Wv + bv), stored TRANSPOSED: vh[bh][d][t] (bf16)
// ---------------------------------------------------------------------------
__global__ void v_proj(const float* __restrict__ vin,
                       const __bf16* __restrict__ wv_t,
                       const float* __restrict__ bv,
                       __bf16* __restrict__ vh) {
  const int wid  = threadIdx.x >> 5;
  const int lane = threadIdx.x & 31;
  const int tile = blockIdx.x * 4 + wid;
  const int ctile = tile & 31;
  const int ttile = tile >> 5;
  const int b = blockIdx.y;

  const int m    = lane & 15;
  const int half = lane >> 4;
  const int tbase = ttile * 16;
  const int cout = ctile * 16 + m;
  const float* arow = vin + ((size_t)(b * Tn + tbase + m)) * Dm;

  f32x8 acc = {0.f,0.f,0.f,0.f,0.f,0.f,0.f,0.f};
  for (int K0 = 0; K0 < Dm; K0 += 32) {
    Frag a;
    const float* p = arow + K0 + half * 8;
    float4 f0 = *(const float4*)p;
    float4 f1 = *(const float4*)(p + 4);
    float4 f2 = *(const float4*)(p + 16);
    float4 f3 = *(const float4*)(p + 20);
    pack8(a, 0, f0, f1);
    pack8(a, 8, f2, f3);
    Frag bf;
    const U4* bp = (const U4*)(wv_t + (size_t)cout * Dm + K0 + half * 8);
    bf.q[0] = bp[0]; bf.q[1] = bp[2];
    acc = wmma_bf16(a, bf, acc);
  }

  const float bi = bv[cout];
  const int d  = cout >> 3;
  const int hh = cout & 7;
  // 8 consecutive t per lane -> one 16B store
  union { __bf16 b[8]; U4 q; } pk;
#pragma unroll
  for (int r = 0; r < 8; ++r) pk.b[r] = tobf(acc[r] + bi);
  size_t base = ((size_t)((b * Hn + hh) * DKv + d)) * Tn + tbase + 8 * half;
  *(U4*)(vh + base) = pk.q;
}

// ---------------------------------------------------------------------------
// Kernel 3: attn logits = qh . kh^T / 8  -> fp32 straight into d_out attn part
// ---------------------------------------------------------------------------
__global__ void qk_logits(const __bf16* __restrict__ qh,
                          const __bf16* __restrict__ kh,
                          float* __restrict__ attn) {
  const int wid  = threadIdx.x >> 5;
  const int lane = threadIdx.x & 31;
  const int tile = blockIdx.x * 4 + wid;          // 0..4095
  const int qtile = tile >> 6;
  const int ktile = tile & 63;
  const int bh = blockIdx.y;

  const int m    = lane & 15;
  const int half = lane >> 4;
  const int qbase = qtile * 16;
  const int kbase = ktile * 16;

  const __bf16* arow = qh + ((size_t)bh * Tn + qbase + m) * DKv;
  const __bf16* brow = kh + ((size_t)bh * Tn + kbase + m) * DKv;

  f32x8 acc = {0.f,0.f,0.f,0.f,0.f,0.f,0.f,0.f};
#pragma unroll
  for (int K0 = 0; K0 < DKv; K0 += 32) {
    Frag a;
    a.q[0] = *(const U4*)(arow + K0 + half * 8);
    a.q[1] = *(const U4*)(arow + K0 + 16 + half * 8);
    Frag bf;
    const U4* bp = (const U4*)(brow + K0 + half * 8);
    bf.q[0] = bp[0]; bf.q[1] = bp[2];
    acc = wmma_bf16(a, bf, acc);
  }

  float* orow = attn + ((size_t)bh * Tn + qbase + 8 * half) * Tn + kbase + m;
#pragma unroll
  for (int r = 0; r < 8; ++r)
    orow[(size_t)r * Tn] = acc[r] * 0.125f;       // 1/sqrt(64)
}

// ---------------------------------------------------------------------------
// Kernel 4: ctx = softmax(attn) @ vh with ONLINE (flash-style) softmax:
//   single pass over the logits; running per-row (max, sum) kept in-lane
//   (row = lane&15, duplicated across halves, combined via shfl_xor 16);
//   accumulators rescaled via 8 shfl row-broadcasts per k-tile.
//   ctx bf16 [b][t][h*64+d]
// ---------------------------------------------------------------------------
__global__ void pv_kernel(const float* __restrict__ attn,
                          const __bf16* __restrict__ vh,
                          __bf16* __restrict__ ctx) {
  const int wid  = threadIdx.x >> 5;
  const int lane = threadIdx.x & 31;
  const int qtile = blockIdx.x * 4 + wid;         // 0..63
  const int bh = blockIdx.y;
  const int b  = bh >> 3;
  const int hs = bh & 7;

  const int m    = lane & 15;
  const int half = lane >> 4;
  const int qbase = qtile * 16;

  const float* lrow = attn + ((size_t)bh * Tn + qbase + m) * Tn;

  f32x8 acc[4];
#pragma unroll
  for (int jt = 0; jt < 4; ++jt) acc[jt] = (f32x8){0.f,0.f,0.f,0.f,0.f,0.f,0.f,0.f};

  float run_m = -INFINITY;     // running row max (rows indexed by lane&15)
  float run_l = 0.f;           // running row sum of exp

  for (int kb = 0; kb < Tn; kb += 32) {
    // this lane's 16 logits of its row (two contiguous groups of 8)
    const float* lp = lrow + kb + half * 8;
    float4 f0 = *(const float4*)lp;
    float4 f1 = *(const float4*)(lp + 4);
    float4 f2 = *(const float4*)(lp + 16);
    float4 f3 = *(const float4*)(lp + 20);
    float e[16] = { f0.x,f0.y,f0.z,f0.w, f1.x,f1.y,f1.z,f1.w,
                    f2.x,f2.y,f2.z,f2.w, f3.x,f3.y,f3.z,f3.w };

    float tm = e[0];
#pragma unroll
    for (int i = 1; i < 16; ++i) tm = fmaxf(tm, e[i]);
    tm = fmaxf(tm, __shfl_xor(tm, 16, 32));       // combine lane halves (same row)
    const float nm = fmaxf(run_m, tm);
    const float scale = __expf(run_m - nm);       // 0 on first tile (-inf)

    Frag p;
    float s = 0.f;
#pragma unroll
    for (int i = 0; i < 16; ++i) {
      float pe = __expf(e[i] - nm);
      s += pe;
      p.h[i] = tobf(pe);
    }
    s += __shfl_xor(s, 16, 32);
    run_l = run_l * scale + s;
    run_m = nm;

    // rescale accumulator rows: row of acc[jt][r] is (r + 8*half)
    float fac[8];
#pragma unroll
    for (int r = 0; r < 8; ++r) fac[r] = __shfl(scale, r + 8 * half, 32);
#pragma unroll
    for (int jt = 0; jt < 4; ++jt)
#pragma unroll
      for (int r = 0; r < 8; ++r) acc[jt][r] *= fac[r];

#pragma unroll
    for (int jt = 0; jt < 4; ++jt) {
      const int dcol = jt * 16 + m;
      Frag bf;
      const U4* bp = (const U4*)(vh + ((size_t)(bh * DKv + dcol)) * Tn + kb + half * 8);
      bf.q[0] = bp[0]; bf.q[1] = bp[2];
      acc[jt] = wmma_bf16(p, bf, acc[jt]);
    }
  }

  // final normalization: 1/l broadcast per accumulator row
  const float invl = 1.f / run_l;
  float inv_r[8];
#pragma unroll
  for (int r = 0; r < 8; ++r) inv_r[r] = __shfl(invl, r + 8 * half, 32);

#pragma unroll
  for (int jt = 0; jt < 4; ++jt) {
    const int cout = hs * DKv + jt * 16 + m;
#pragma unroll
    for (int r = 0; r < 8; ++r) {
      int t = qbase + r + 8 * half;
      ctx[((size_t)(b * Tn + t)) * Dm + cout] = tobf(acc[jt][r] * inv_r[r]);
    }
  }
}

// ---------------------------------------------------------------------------
// Kernel 5: out = LayerNorm( relu(ctx @ Wfc + bfc) + residual )
//   256 threads = 8 waves; the shared 16-row A tile (16 KB, contiguous in ctx)
//   is staged once through LDS (8x traffic reduction: all waves use the same A);
//   each wave owns a 64-col slice (4 WMMA tiles); fused LN via LDS + shuffles.
// ---------------------------------------------------------------------------
#define LDSS 516
__global__ void fc_ln(const __bf16* __restrict__ ctx,
                      const __bf16* __restrict__ wfc_t,
                      const float* __restrict__ bfc,
                      const float* __restrict__ qres,
                      const float* __restrict__ gamma, const float* __restrict__ beta,
                      float* __restrict__ out) {
  __shared__ float tile[16 * LDSS];
  __shared__ __align__(16) unsigned short aT[16 * Dm];   // 16 KB bf16 A tile

  const int wid  = threadIdx.x >> 5;
  const int lane = threadIdx.x & 31;
  const int m    = lane & 15;
  const int half = lane >> 4;
  const int R = blockIdx.x * 16;                  // global row base over B*T

  // cooperative stage: rows R..R+15 are contiguous 16 KB in ctx
  {
    const U4* src = (const U4*)(ctx + (size_t)R * Dm);
    U4* dst = (U4*)aT;
#pragma unroll
    for (int i = threadIdx.x; i < 16 * Dm / 8; i += 256)
      dst[i] = src[i];
  }
  __syncthreads();

  f32x8 acc[4];
#pragma unroll
  for (int jt = 0; jt < 4; ++jt) acc[jt] = (f32x8){0.f,0.f,0.f,0.f,0.f,0.f,0.f,0.f};

  for (int K0 = 0; K0 < Dm; K0 += 32) {
    Frag a;
    a.q[0] = *(const U4*)(aT + m * Dm + K0 + half * 8);
    a.q[1] = *(const U4*)(aT + m * Dm + K0 + 16 + half * 8);
#pragma unroll
    for (int jt = 0; jt < 4; ++jt) {
      const int cout = wid * 64 + jt * 16 + m;
      Frag bf;
      const U4* bp = (const U4*)(wfc_t + (size_t)cout * Dm + K0 + half * 8);
      bf.q[0] = bp[0]; bf.q[1] = bp[2];
      acc[jt] = wmma_bf16(a, bf, acc[jt]);
    }
  }

#pragma unroll
  for (int jt = 0; jt < 4; ++jt) {
    const int cout = wid * 64 + jt * 16 + m;
    const float bi = bfc[cout];
#pragma unroll
    for (int r = 0; r < 8; ++r) {
      int mr = r + 8 * half;
      float y = fmaxf(acc[jt][r] + bi, 0.f);                 // relu
      y += qres[(size_t)(R + mr) * Dm + cout];               // residual
      tile[mr * LDSS + cout] = y;
    }
  }
  __syncthreads();

  // LayerNorm: each wave handles 2 rows, 16 lanes per row.
  const int mrow = wid * 2 + (lane >> 4);
  const int cl = lane & 15;
  float s = 0.f, sq = 0.f;
  for (int j = cl; j < Dm; j += 16) {
    float x = tile[mrow * LDSS + j];
    s += x; sq += x * x;
  }
#pragma unroll
  for (int o = 8; o; o >>= 1) {
    s  += __shfl_xor(s,  o, 32);
    sq += __shfl_xor(sq, o, 32);
  }
  const float mean = s * (1.f / Dm);
  float var = sq * (1.f / Dm) - mean * mean;
  var = fmaxf(var, 0.f);
  const float rstd = rsqrtf(var + LN_EPS);

  float* orow = out + (size_t)(R + mrow) * Dm;
  for (int j = cl; j < Dm; j += 16)
    orow[j] = (tile[mrow * LDSS + j] - mean) * rstd * gamma[j] + beta[j];
}

// ---------------------------------------------------------------------------
extern "C" void kernel_launch(void* const* d_in, const int* in_sizes, int n_in,
                              void* d_out, int out_size, void* d_ws, size_t ws_size,
                              hipStream_t stream) {
  (void)in_sizes; (void)n_in; (void)out_size; (void)ws_size;

  const float* q    = (const float*)d_in[0];
  const float* k    = (const float*)d_in[1];
  const float* v    = (const float*)d_in[2];
  const float* Wq   = (const float*)d_in[3];
  const float* bq   = (const float*)d_in[4];
  const float* Wk   = (const float*)d_in[5];
  const float* bk   = (const float*)d_in[6];
  const float* Wv   = (const float*)d_in[7];
  const float* bv   = (const float*)d_in[8];
  const float* Wfc  = (const float*)d_in[9];
  const float* bfc  = (const float*)d_in[10];
  const float* gamma= (const float*)d_in[11];
  const float* beta = (const float*)d_in[12];

  float* outp  = (float*)d_out;                 // (B,T,512)
  float* attnp = outp + OUT0;                   // (B,H,T,T) pre-softmax logits

  char* w = (char*)d_ws;
  __bf16* wq_t = (__bf16*)(w);                       // 1.5 MB
  __bf16* wk_t = (__bf16*)(w + 1572864);             // 1.5 MB
  __bf16* wv_t = (__bf16*)(w + 3145728);             // 0.5 MB
  __bf16* wfc_t= (__bf16*)(w + 3670016);             // 0.5 MB
  __bf16* qh   = (__bf16*)(w + 4194304);             // 8 MB
  __bf16* kh   = (__bf16*)(w + 12582912);            // 8 MB
  __bf16* vh   = (__bf16*)(w + 20971520);            // 8 MB
  __bf16* ctx  = (__bf16*)(w + 29360128);            // 8 MB
  // total ~36 MB

  prep_weights<<<dim3(3072), dim3(256), 0, stream>>>(Wq, Wk, Wv, Wfc,
                                                     wq_t, wk_t, wv_t, wfc_t);

  qk_conv_proj<<<dim3(512, Bn, 2), dim3(128), 0, stream>>>(q, k, wq_t, wk_t,
                                                           bq, bk, qh, kh);

  v_proj<<<dim3(512, Bn), dim3(128), 0, stream>>>(v, wv_t, bv, vh);

  qk_logits<<<dim3(1024, Bn * Hn), dim3(128), 0, stream>>>(qh, kh, attnp);

  pv_kernel<<<dim3(16, Bn * Hn), dim3(128), 0, stream>>>(attnp, vh, ctx);

  fc_ln<<<dim3(NROWS / 16), dim3(256), 0, stream>>>(ctx, wfc_t, bfc, q,
                                                    gamma, beta, outp);
}
